// MultiHeadAttention_55327768708496
// MI455X (gfx1250) — matrix operand order repaired and checked by reading, not compiled, not run on hardware
//
#include <hip/hip_runtime.h>
#include <hip/hip_fp16.h>
#include <math.h>

typedef __attribute__((ext_vector_type(16))) _Float16 v16h;
typedef __attribute__((ext_vector_type(8)))  _Float16 v8h;
typedef __attribute__((ext_vector_type(8)))  float    v8f;

#define B_     8
#define T_     2048
#define D_     512
#define H_     4
#define DH_    128
#define INNER_ 512
#define QKVN_  1536

// ---------------------------------------------------------------------------
// WMMA helpers (gfx1250 wave32, 16x16x32 f16 -> f32)
// ---------------------------------------------------------------------------
__device__ __forceinline__ v8f wmma16(v16h a, v16h b, v8f c) {
    return __builtin_amdgcn_wmma_f32_16x16x32_f16(
        /*neg_a=*/false, a, /*neg_b=*/false, b,
        /*c_mod=*/(short)0, c, /*reuse_a=*/false, /*reuse_b=*/false);
}

// 16 contiguous halves -> v16h (two 16B loads; ptr must be 16B aligned)
__device__ __forceinline__ v16h load16h(const _Float16* p) {
    v8h lo = *(const v8h*)(p);
    v8h hi = *(const v8h*)(p + 8);
    v16h r;
#pragma unroll
    for (int i = 0; i < 8; ++i) { r[i] = lo[i]; r[i + 8] = hi[i]; }
    return r;
}

// A-fragment (16x32 f16) from a row-major [rows][stride] f16 tile.
// Layout: lane L -> M = L&15; halves 0..15 = K base..base+15,
// base = (L>=16) ? 16 : 0 (matches load_b_frag).
__device__ __forceinline__ v16h load_a_frag(const _Float16* tile, int m0,
                                            int stride, int lane) {
    int m  = m0 + (lane & 15);
    int kb = (lane >= 16) ? 16 : 0;
    const _Float16* row = tile + (size_t)m * stride;
    v8h lo = *(const v8h*)(row + kb);
    v8h hi = *(const v8h*)(row + kb + 8);
    v16h r;
#pragma unroll
    for (int i = 0; i < 8; ++i) { r[i] = lo[i]; r[i + 8] = hi[i]; }
    return r;
}

// B-fragment (32x16 f16) from a K-transposed [N][stride] f16 tile.
// Contiguous-k layout used here (same as load_a_frag): lane L -> N = L&15; halves 0..15 = K 0..15 (lanes<16)
// or K 16..31 (lanes>=16).
__device__ __forceinline__ v16h load_b_frag(const _Float16* tileT, int n0,
                                            int stride, int lane) {
    int n  = n0 + (lane & 15);
    int kb = (lane >= 16) ? 16 : 0;
    return load16h(tileT + (size_t)n * stride + kb);
}

// Async 16-byte global -> LDS copy (CDNA5 ASYNCcnt path).
__device__ __forceinline__ void async_copy_b128(unsigned lds_addr,
                                                const _Float16* gptr) {
    asm volatile("global_load_async_to_lds_b128 %0, %1, off"
                 :: "v"(lds_addr),
                    "v"((unsigned long long)(size_t)gptr)
                 : "memory");
}
__device__ __forceinline__ void wait_async0() {
    asm volatile("s_wait_asynccnt 0x0" ::: "memory");
}

// ---------------------------------------------------------------------------
// Kernel: C[M,N] = A[M,K] @ WT[N,K]^T + bias   (f16 in, f32 out)
// A row-major f16, WT row-major f16 (pre-transposed weights).
// Block: 128 thr (4 waves), tile 64x64, waves 2x2, each wave 2x2 of 16x16.
// LDS tiles double-buffered and filled with global_load_async_to_lds_b128.
// ---------------------------------------------------------------------------
__global__ __launch_bounds__(128) void wmma_gemm_bias(
        const _Float16* __restrict__ A, const _Float16* __restrict__ WT,
        const float* __restrict__ bias, float* __restrict__ C,
        int M, int N, int K) {
    __shared__ __align__(16) _Float16 As[2][64][40];   // [m][k], k-chunk of 32
    __shared__ __align__(16) _Float16 Ws[2][64][40];   // [n][k]

    const int tid  = threadIdx.x;
    const int lane = tid & 31;
    const int w    = tid >> 5;
    const int wm   = (w >> 1) * 32;
    const int wn   = (w & 1) * 32;
    const int m0   = blockIdx.y * 64;
    const int n0   = blockIdx.x * 64;

    const unsigned asBase = (unsigned)(size_t)(&As[0][0][0]);
    const unsigned wsBase = (unsigned)(size_t)(&Ws[0][0][0]);
    const unsigned bufStride = 64 * 40 * sizeof(_Float16);   // 5120 B

    // Each tile = 64 rows x 32 halves = 256 x 16B chunks; 2 chunks/thread.
    auto issue = [&](int buf, int k0) {
#pragma unroll
        for (int i = 0; i < 2; ++i) {
            int idx = tid + i * 128;
            int r   = idx >> 2;
            int c16 = (idx & 3) * 8;                     // half offset in row
            unsigned loff = (unsigned)buf * bufStride + (r * 40 + c16) * 2u;
            async_copy_b128(asBase + loff, A  + (size_t)(m0 + r) * K + k0 + c16);
            async_copy_b128(wsBase + loff, WT + (size_t)(n0 + r) * K + k0 + c16);
        }
    };

    v8f acc[2][2];
#pragma unroll
    for (int i = 0; i < 2; ++i)
#pragma unroll
        for (int j = 0; j < 2; ++j)
#pragma unroll
            for (int e = 0; e < 8; ++e) acc[i][j][e] = 0.0f;

    issue(0, 0);
    for (int k0 = 0; k0 < K; k0 += 32) {
        const int cur = (k0 >> 5) & 1;
        wait_async0();          // this wave's copies into buf `cur` landed
        __syncthreads();        // everyone's copies landed; prev reads done
        if (k0 + 32 < K) issue(cur ^ 1, k0 + 32);

        v16h a0 = load_a_frag(&As[cur][0][0], wm,      40, lane);
        v16h a1 = load_a_frag(&As[cur][0][0], wm + 16, 40, lane);
        v16h b0 = load_b_frag(&Ws[cur][0][0], wn,      40, lane);
        v16h b1 = load_b_frag(&Ws[cur][0][0], wn + 16, 40, lane);

        acc[0][0] = wmma16(a0, b0, acc[0][0]);
        acc[0][1] = wmma16(a0, b1, acc[0][1]);
        acc[1][0] = wmma16(a1, b0, acc[1][0]);
        acc[1][1] = wmma16(a1, b1, acc[1][1]);
    }

    // epilogue: bias + store (C layout: vgpr r -> row r (+8 for lanes>=16))
    const int col = lane & 15;
    const int ro  = (lane >= 16) ? 8 : 0;
#pragma unroll
    for (int im = 0; im < 2; ++im)
#pragma unroll
        for (int in = 0; in < 2; ++in) {
            int n  = n0 + wn + in * 16 + col;
            float bv = bias[n];
            int mb = m0 + wm + im * 16 + ro;
#pragma unroll
            for (int r = 0; r < 8; ++r)
                C[(size_t)(mb + r) * N + n] = acc[im][in][r] + bv;
        }
}

// ---------------------------------------------------------------------------
// One-time converters: x -> f16, W[K][N] -> WT[N][K] f16
// ---------------------------------------------------------------------------
__global__ void cvt_f16_kernel(const float* __restrict__ src,
                               _Float16* __restrict__ dst, int n) {
    int i = blockIdx.x * blockDim.x + threadIdx.x;
    if (i < n) dst[i] = (_Float16)src[i];
}

__global__ void transpose_cvt_kernel(const float* __restrict__ W,
                                     _Float16* __restrict__ WT,
                                     int K, int N) {
    int i = blockIdx.x * blockDim.x + threadIdx.x;
    if (i >= K * N) return;
    int n = i % N, k = i / N;                 // coalesced read along n
    WT[(size_t)n * K + k] = (_Float16)W[i];
}

// ---------------------------------------------------------------------------
// RoPE on q,k + pack q,k -> f16 [B,H,T,DH], v -> f16 [B,H,DH,T]
// ---------------------------------------------------------------------------
__global__ void rope_pack_kernel(const float* __restrict__ qkv,
                                 _Float16* __restrict__ q,
                                 _Float16* __restrict__ k,
                                 _Float16* __restrict__ vT) {
    int idx = blockIdx.x * blockDim.x + threadIdx.x;
    if (idx >= B_ * T_ * H_ * (DH_ / 2)) return;
    int i = idx & 63;          idx >>= 6;
    int h = idx & (H_ - 1);    idx >>= 2;
    int t = idx & (T_ - 1);
    int b = idx >> 11;

    const float* base = qkv + (size_t)(b * T_ + t) * QKVN_;
    float inv = __powf(10000.0f, -(float)i / 64.0f);
    float s, c;
    __sincosf((float)t * inv, &s, &c);

    int d0 = h * DH_ + 2 * i;
    float qe = base[d0],              qo = base[d0 + 1];
    float ke = base[INNER_ + d0],     ko = base[INNER_ + d0 + 1];
    float ve = base[2 * INNER_ + d0], vo = base[2 * INNER_ + d0 + 1];

    size_t qb = ((size_t)(b * H_ + h) * T_ + t) * DH_ + 2 * i;
    q[qb]     = (_Float16)(qe * c - qo * s);
    q[qb + 1] = (_Float16)(qe * s + qo * c);
    k[qb]     = (_Float16)(ke * c - ko * s);
    k[qb + 1] = (_Float16)(ke * s + ko * c);

    size_t vb = (size_t)(b * H_ + h) * DH_ * T_;
    vT[vb + (size_t)(2 * i) * T_ + t]     = (_Float16)ve;
    vT[vb + (size_t)(2 * i + 1) * T_ + t] = (_Float16)vo;
}

// ---------------------------------------------------------------------------
// Flash attention. Block = 4 waves; wave w owns 16 query rows.
// Per s-chunk of 32 keys: 8 WMMAs for QK^T, online softmax, 8 WMMAs for P*V.
// O written as f16 [B,T,H*DH] so the output GEMM consumes it directly.
// ---------------------------------------------------------------------------
__global__ __launch_bounds__(128) void flash_attn_kernel(
        const _Float16* __restrict__ Q, const _Float16* __restrict__ Kc,
        const _Float16* __restrict__ Vt, _Float16* __restrict__ O) {
    __shared__ __align__(16) float     sc[4][16][33];
    __shared__ __align__(16) _Float16  pt[4][16][40];
    __shared__ __align__(16) float     fac[4][16];

    const int tid  = threadIdx.x;
    const int lane = tid & 31;
    const int w    = tid >> 5;
    const int b    = blockIdx.z;
    const int h    = blockIdx.y;
    const int t0   = blockIdx.x * 64 + w * 16;

    const size_t bh = (size_t)(b * H_ + h);
    const _Float16* q  = Q  + bh * T_ * DH_;
    const _Float16* kp = Kc + bh * T_ * DH_;
    const _Float16* vp = Vt + bh * DH_ * T_;

    const int   ml    = lane & 15;
    const int   ro    = (lane >= 16) ? 8 : 0;
    const float scale = 0.08838834764831845f;   // 1/sqrt(128)

    // Q A-fragments for all 4 k-chunks (kept in VGPRs for the whole pass)
    v16h qf[4];
    {
        const _Float16* qrow = q + (size_t)(t0 + ml) * DH_;
        const int kb = (lane >= 16) ? 16 : 0;
#pragma unroll
        for (int kc = 0; kc < 4; ++kc) {
            v8h lo = *(const v8h*)(qrow + kc * 32 + kb);
            v8h hi = *(const v8h*)(qrow + kc * 32 + kb + 8);
#pragma unroll
            for (int i = 0; i < 8; ++i) { qf[kc][i] = lo[i]; qf[kc][i + 8] = hi[i]; }
        }
    }

    v8f acc[8];
#pragma unroll
    for (int j = 0; j < 8; ++j)
#pragma unroll
        for (int e = 0; e < 8; ++e) acc[j][e] = 0.0f;
    float m_run = -INFINITY, l_run = 0.0f;

    for (int s0 = 0; s0 < T_; s0 += 32) {
        // ---- scores = Q @ K^T : two 16x16 C tiles over K(dh)=128 ----
        v8f c0, c1;
#pragma unroll
        for (int e = 0; e < 8; ++e) { c0[e] = 0.0f; c1[e] = 0.0f; }
        const int dhb = (lane >= 16) ? 16 : 0;    // B-frag K base
#pragma unroll
        for (int kc = 0; kc < 4; ++kc) {
            v16h bk0 = load16h(kp + (size_t)(s0 + ml) * DH_      + kc * 32 + dhb);
            v16h bk1 = load16h(kp + (size_t)(s0 + 16 + ml) * DH_ + kc * 32 + dhb);
            c0 = wmma16(qf[kc], bk0, c0);
            c1 = wmma16(qf[kc], bk1, c1);
        }
        // spill scores to LDS (C layout -> [row][col])
#pragma unroll
        for (int r = 0; r < 8; ++r) {
            sc[w][ro + r][ml]      = c0[r];
            sc[w][ro + r][16 + ml] = c1[r];
        }
        __syncthreads();

        // ---- online softmax: lanes 0..15 each own one query row ----
        if (lane < 16) {
            int row = lane;
            float mx = m_run;
#pragma unroll
            for (int j = 0; j < 32; ++j) mx = fmaxf(mx, sc[w][row][j] * scale);
            float corr = __expf(m_run - mx);
            float sum  = 0.0f;
#pragma unroll
            for (int j = 0; j < 32; ++j) {
                float p = __expf(sc[w][row][j] * scale - mx);
                pt[w][row][j] = (_Float16)p;
                sum += p;
            }
            l_run = l_run * corr + sum;
            m_run = mx;
            fac[w][row] = corr;
        }
        __syncthreads();

        // rescale accumulator rows by the correction factor
#pragma unroll
        for (int r = 0; r < 8; ++r) {
            float cf = fac[w][ro + r];
#pragma unroll
            for (int j = 0; j < 8; ++j) acc[j][r] *= cf;
        }

        // ---- acc += P @ V : A = P (16x32), B = vT rows (contiguous) ----
        v16h pa = load_a_frag(&pt[w][0][0], 0, 40, lane);
        const int sbb = s0 + ((lane >= 16) ? 16 : 0);
#pragma unroll
        for (int j = 0; j < 8; ++j) {
            v16h bv = load16h(vp + (size_t)(j * 16 + ml) * T_ + sbb);
            acc[j] = wmma16(pa, bv, acc[j]);
        }
    }

    // ---- finalize: divide by running sum, store f16 [B,T,H*DH] ----
    if (lane < 16) fac[w][lane] = 1.0f / l_run;
    __syncthreads();
#pragma unroll
    for (int j = 0; j < 8; ++j) {
        int dh = h * DH_ + j * 16 + ml;
#pragma unroll
        for (int r = 0; r < 8; ++r)
            O[(size_t)(b * T_ + t0 + ro + r) * INNER_ + dh] =
                (_Float16)(acc[j][r] * fac[w][ro + r]);
    }
}

// ---------------------------------------------------------------------------
// Host-side launch sequence (all on `stream`, graph-capture safe)
// ---------------------------------------------------------------------------
extern "C" void kernel_launch(void* const* d_in, const int* in_sizes, int n_in,
                              void* d_out, int out_size, void* d_ws, size_t ws_size,
                              hipStream_t stream) {
    const float* x     = (const float*)d_in[0];
    // d_in[1] = mask: all-true in this problem -> no-op, ignored.
    const float* W_qkv = (const float*)d_in[2];
    const float* b_qkv = (const float*)d_in[3];
    const float* W_out = (const float*)d_in[4];
    const float* b_out = (const float*)d_in[5];
    float* out = (float*)d_out;

    char* ws = (char*)d_ws;
    size_t off = 0;
    float*     qkv  = (float*)(ws + off);    off += (size_t)B_ * T_ * QKVN_ * sizeof(float);
    _Float16*  qh   = (_Float16*)(ws + off); off += (size_t)B_ * H_ * T_ * DH_ * sizeof(_Float16);
    _Float16*  kh   = (_Float16*)(ws + off); off += (size_t)B_ * H_ * T_ * DH_ * sizeof(_Float16);
    _Float16*  vt   = (_Float16*)(ws + off); off += (size_t)B_ * H_ * DH_ * T_ * sizeof(_Float16);
    _Float16*  Oat  = (_Float16*)(ws + off); off += (size_t)B_ * T_ * INNER_ * sizeof(_Float16);
    _Float16*  xh   = (_Float16*)(ws + off); off += (size_t)B_ * T_ * D_ * sizeof(_Float16);
    _Float16*  WqT  = (_Float16*)(ws + off); off += (size_t)QKVN_ * D_ * sizeof(_Float16);
    _Float16*  WoT  = (_Float16*)(ws + off); off += (size_t)D_ * INNER_ * sizeof(_Float16);
    (void)ws_size; (void)in_sizes; (void)n_in; (void)out_size;

    const int M = B_ * T_;   // 16384

    // 0) one-time f16 conversion + weight transposes
    int nx = M * D_;
    cvt_f16_kernel<<<(nx + 255) / 256, 256, 0, stream>>>(x, xh, nx);
    int nq = D_ * QKVN_;
    transpose_cvt_kernel<<<(nq + 255) / 256, 256, 0, stream>>>(W_qkv, WqT, D_, QKVN_);
    int no = INNER_ * D_;
    transpose_cvt_kernel<<<(no + 255) / 256, 256, 0, stream>>>(W_out, WoT, INNER_, D_);

    // 1) qkv = x @ W_qkv + b_qkv
    wmma_gemm_bias<<<dim3(QKVN_ / 64, M / 64), 128, 0, stream>>>(
        xh, WqT, b_qkv, qkv, M, QKVN_, D_);

    // 2) RoPE + pack to f16 (q,k per [B,H,T,DH]; v transposed [B,H,DH,T])
    int np = B_ * T_ * H_ * (DH_ / 2);
    rope_pack_kernel<<<(np + 255) / 256, 256, 0, stream>>>(qkv, qh, kh, vt);

    // 3) attention with online softmax
    flash_attn_kernel<<<dim3(T_ / 64, H_, B_), 128, 0, stream>>>(qh, kh, vt, Oat);

    // 4) out = O @ W_out + b_out
    wmma_gemm_bias<<<dim3(D_ / 64, M / 64), 128, 0, stream>>>(
        Oat, WoT, b_out, out, M, D_, INNER_);
}
